// AR_RNN_GRU_11836929868327
// MI455X (gfx1250) — matrix operand order repaired and checked by reading, not compiled, not run on hardware
//
#include <hip/hip_runtime.h>
#include <stdint.h>

// ---------------- problem constants ----------------
#define B_     32
#define T_IN_  256
#define T_OUT_ 256
#define D_     128
#define U_     256
#define NSTEP  511          // 256 warmup + 255 autoregressive stack steps
#define NWG    16           // persistent workgroups (16 jtiles); 2 waves each (mtile)

// LDS weight cache layout (units of 1KB "fragment lines" = 256 u32 = 32 lanes x 32B)
#define L_Z0   0
#define L_R0   12
#define L_XN0  24
#define L_HN0  28
#define L_Z1   36
#define L_R1   52
#define L_XN1  68
#define L_HN1  76
#define L_Z2   84
#define L_R2   100
#define L_XN2  116
#define L_HN2  124
#define L_DEN  132
#define L_TOT  140          // 140 KB of weight fragments per workgroup

typedef __attribute__((ext_vector_type(16))) __bf16      v16bf;
typedef __attribute__((ext_vector_type(8)))  float       v8f;
typedef __attribute__((ext_vector_type(8)))  unsigned    v8u;

// ---------------- helpers ----------------
__device__ inline unsigned short f2bf(float f) {
  unsigned u = __float_as_uint(f);
  u += 0x7FFFu + ((u >> 16) & 1u);   // RNE
  return (unsigned short)(u >> 16);
}
__device__ inline float sigm(float x)  { return 1.0f / (1.0f + __expf(-x)); }
__device__ inline float tanhf_(float x){ return 1.0f - 2.0f / (__expf(2.0f * x) + 1.0f); }

// A-fragment (16-bit A 16x32, ISA 7.12.2): lane holds row m=lane&15, K-half by lane>>4.
// act is row-major bf16 [32][rowW]; u32 cols {16*kc + hi*4 + 0..3} and {+8}.
__device__ inline v16bf afrag(const unsigned short* act, int rowW, int m, int kc, int hi) {
  const unsigned* p = (const unsigned*)(act + (size_t)m * rowW) + kc * 16 + hi * 4;
  uint4 a = *(const uint4*)p;
  uint4 b = *(const uint4*)(p + 8);
  v8u r; r[0]=a.x; r[1]=a.y; r[2]=a.z; r[3]=a.w; r[4]=b.x; r[5]=b.y; r[6]=b.z; r[7]=b.w;
  return __builtin_bit_cast(v16bf, r);
}
// B-fragment from the LDS weight cache: line = lineBase+kc, lane reads 32B.
__device__ inline v16bf bfrag_s(const unsigned* sW, int lineBase, int kc, int lane) {
  const v8u* p = (const v8u*)(sW + ((lineBase + kc) << 8) + (lane << 3));
  return __builtin_bit_cast(v16bf, *p);
}
__device__ inline v8f wmma_bf16(v16bf a, v16bf b, v8f c) {
  return __builtin_amdgcn_wmma_f32_16x16x32_bf16(false, a, false, b, (short)0, c, false, false);
}

// ---------------- parameter block ----------------
struct KP {
  unsigned short* xin;                       // [T_IN][32][128] bf16 (inputs+noise)
  unsigned *zf0,*rf0,*xnf0,*hnf0;            // layer0 fragments (KC=12, KX=4)
  unsigned *zf1,*rf1,*xnf1,*hnf1;            // layer1 (KC=16, KX=8)
  unsigned *zf2,*rf2,*xnf2,*hnf2;            // layer2
  unsigned *dfrag;                           // dense  (8 jt x 8 kc)
  float *bzr0,*bxn0,*bhn0;
  float *bzr1,*bxn1,*bhn1;
  float *bzr2,*bxn2,*bhn2;
  float *bd;
  float *h0F,*h1F,*h2F,*x2f;                 // f32 [32][256]
  unsigned short *h0p[2],*h1p[2],*h2p[2];    // bf16 packs, double buffered
  unsigned short *x2p,*x3p,*predp;           // bf16 [32][256],[32][256],[32][128]
  unsigned *bcnt,*bgen;
  float *out;                                // [32][256][128] f32
};

// ---------------- prep kernels ----------------
__global__ void k_pack_xin(const float* __restrict__ inp, const float* __restrict__ noi,
                           unsigned short* __restrict__ xin, int total) {
  int i = blockIdx.x * blockDim.x + threadIdx.x;
  if (i >= total) return;
  int d = i % D_; int rest = i / D_; int b = rest % B_; int t = rest / B_;
  size_t s = ((size_t)b * T_IN_ + t) * D_ + d;
  xin[i] = f2bf(inp[s] + noi[s]);
}

// Pack one GRU layer's weights into WMMA B fragment order (per-jtile contiguous).
// z/r gates: concat [W;U] over K (KC chunks); xn: W cols 512+..; hn: U cols 512+..
__global__ void k_pack_gru(const float* __restrict__ W, const float* __restrict__ Ur, int Dl,
                           unsigned* __restrict__ zf, unsigned* __restrict__ rf,
                           unsigned* __restrict__ xnf, unsigned* __restrict__ hnf, int total) {
  int i = blockIdx.x * blockDim.x + threadIdx.x;
  if (i >= total) return;
  int KX = Dl / 32, KC = KX + 8;
  int szZR = 16 * KC * 256;
  int szXN = 16 * KX * 256;
  int i2, KCc, colOff, sect;
  if (i < szZR)               { i2 = i;                KCc = KC; colOff = 0;   sect = 0; }
  else if (i < 2 * szZR)      { i2 = i - szZR;         KCc = KC; colOff = 256; sect = 1; }
  else if (i < 2*szZR + szXN) { i2 = i - 2 * szZR;     KCc = KX; colOff = 512; sect = 2; }
  else                        { i2 = i - 2*szZR - szXN;KCc = 8;  colOff = 512; sect = 3; }
  int v = i2 & 7, lane = (i2 >> 3) & 31, rest = i2 >> 8;
  int kc = rest % KCc, jt = rest / KCc;
  int koff = 2 * (v & 3) + 8 * (lane >> 4) + 16 * (v >> 2);  // 16-bit B layout: k = 8*hi + s (s<8), 16 + 8*hi + (s-8)
  int k = kc * 32 + koff;
  int col = jt * 16 + (lane & 15) + colOff;
  float a, b;
  if (sect <= 1) {
    if (k < Dl) { a = W[(size_t)k * 768 + col];         b = W[(size_t)(k + 1) * 768 + col]; }
    else        { a = Ur[(size_t)(k - Dl) * 768 + col]; b = Ur[(size_t)(k - Dl + 1) * 768 + col]; }
  } else if (sect == 2) { a = W[(size_t)k * 768 + col];  b = W[(size_t)(k + 1) * 768 + col]; }
  else                  { a = Ur[(size_t)k * 768 + col]; b = Ur[(size_t)(k + 1) * 768 + col]; }
  unsigned pv = (unsigned)f2bf(a) | ((unsigned)f2bf(b) << 16);
  unsigned* dst = (sect == 0) ? zf : (sect == 1) ? rf : (sect == 2) ? xnf : hnf;
  dst[i2] = pv;
}

__global__ void k_pack_dense(const float* __restrict__ Wd, unsigned* __restrict__ dfrag, int total) {
  int i = blockIdx.x * blockDim.x + threadIdx.x;
  if (i >= total) return;
  int v = i & 7, lane = (i >> 3) & 31, rest = i >> 8;
  int kc = rest % 8, jt = rest / 8;
  int koff = 2 * (v & 3) + 8 * (lane >> 4) + 16 * (v >> 2);
  int k = kc * 32 + koff;
  int col = jt * 16 + (lane & 15);
  unsigned pv = (unsigned)f2bf(Wd[(size_t)k * 128 + col]) |
                ((unsigned)f2bf(Wd[(size_t)(k + 1) * 128 + col]) << 16);
  dfrag[i] = pv;
}

// states + fused biases + barrier reset
__global__ void k_init(KP P, const float* __restrict__ h_init,
                       const float* __restrict__ b0, const float* __restrict__ b1,
                       const float* __restrict__ b2, const float* __restrict__ bdIn) {
  int i = blockIdx.x * blockDim.x + threadIdx.x;
  if (i < 3 * B_ * U_) {                                     // hidden states
    int l = i / (B_ * U_); int r = i % (B_ * U_);
    float val = h_init[l * U_ + (r % U_)];
    float* hF = (l == 0) ? P.h0F : (l == 1) ? P.h1F : P.h2F;
    unsigned short* p0 = (l == 0) ? P.h0p[0] : (l == 1) ? P.h1p[0] : P.h2p[0];
    unsigned short* p1 = (l == 0) ? P.h0p[1] : (l == 1) ? P.h1p[1] : P.h2p[1];
    hF[r] = val; unsigned short us = f2bf(val); p0[r] = us; p1[r] = us;
    return;
  }
  int j = i - 3 * B_ * U_;
  if (j < 3 * 1024) {                                        // fused biases
    int l = j / 1024, c = j % 1024;
    const float* bl = (l == 0) ? b0 : (l == 1) ? b1 : b2;    // [2][768] flat
    float* bzr = (l == 0) ? P.bzr0 : (l == 1) ? P.bzr1 : P.bzr2;
    float* bxn = (l == 0) ? P.bxn0 : (l == 1) ? P.bxn1 : P.bxn2;
    float* bhn = (l == 0) ? P.bhn0 : (l == 1) ? P.bhn1 : P.bhn2;
    if (c < 512)      bzr[c] = bl[c] + bl[768 + c];          // z,r: input+recurrent bias
    else if (c < 768) bxn[c - 512] = bl[512 + (c - 512)];    // n input bias
    else              bhn[c - 768] = bl[1280 + (c - 768)];   // n recurrent bias
    return;
  }
  j -= 3 * 1024;
  if (j < D_) { P.bd[j] = bdIn[j]; return; }
  if (j == D_)     *P.bcnt = 0u;
  if (j == D_ + 1) *P.bgen = 0u;
}

// ---------------- device-wide barrier ----------------
__device__ inline void gbar(unsigned* cnt, unsigned* gen, unsigned& ep) {
  __threadfence();
  __syncthreads();
  if (threadIdx.x == 0) {
    ++ep;
    unsigned old = __hip_atomic_fetch_add(cnt, 1u, __ATOMIC_ACQ_REL, __HIP_MEMORY_SCOPE_AGENT);
    if (old == NWG - 1u) {
      __hip_atomic_store(cnt, 0u, __ATOMIC_RELAXED, __HIP_MEMORY_SCOPE_AGENT);
      __hip_atomic_store(gen, ep, __ATOMIC_RELEASE, __HIP_MEMORY_SCOPE_AGENT);
    } else {
      while (__hip_atomic_load(gen, __ATOMIC_ACQUIRE, __HIP_MEMORY_SCOPE_AGENT) < ep)
        __builtin_amdgcn_s_sleep(1);
    }
  }
  __syncthreads();
}

// ---------------- GRU layer phase (one wave = one (jtile, mtile)) ----------------
// Weights + biases come from the LDS cache (sW fragment lines, sB per-layer 64-float block).
__device__ void gru_layer(int mt, int lane,
                          const unsigned short* xsrc, int xw,       // x activation pack
                          const unsigned short* hR, int KX,         // recurrent pack (read)
                          const unsigned* sW, int zL, int rL, int xnL, int hnL,
                          const float* sB,                          // [bz16|br16|bx16|bh16]
                          int col,                                  // global column
                          float* hF, unsigned short* hW,            // h state f32 + pack (write)
                          const float* resIn, float* resOutF, unsigned short* resOutP) {
  const int hi = lane >> 4, nl = lane & 15, m0 = mt * 16;
  const int KC = KX + 8;
  v8f az = {}, arr = {}, axn = {}, ahn = {};
  for (int kc = 0; kc < KC; ++kc) {
    v16bf a = (kc < KX) ? afrag(xsrc, xw, m0 + nl, kc, hi)
                        : afrag(hR, U_, m0 + nl, kc - KX, hi);
    az  = wmma_bf16(a, bfrag_s(sW, zL, kc, lane), az);
    arr = wmma_bf16(a, bfrag_s(sW, rL, kc, lane), arr);
    if (kc < KX) axn = wmma_bf16(a, bfrag_s(sW, xnL, kc, lane), axn);
    else         ahn = wmma_bf16(a, bfrag_s(sW, hnL, kc - KX, lane), ahn);
  }
  const float bz = sB[nl], br = sB[16 + nl], bx = sB[32 + nl], bh = sB[48 + nl];
#pragma unroll
  for (int r = 0; r < 8; ++r) {                 // C/D f32 layout: m = r + 8*hi
    int m = m0 + r + 8 * hi;
    float z  = sigm(az[r] + bz);
    float rg = sigm(arr[r] + br);
    float n  = tanhf_(axn[r] + bx + rg * (ahn[r] + bh));
    int idx = m * U_ + col;
    float hnew = z * hF[idx] + (1.0f - z) * n;
    hF[idx] = hnew;
    hW[idx] = f2bf(hnew);
    if (resOutF) {                               // layer1: x2 = h1 + x1 (f32 + pack)
      float xr = hnew + resIn[idx];
      resOutF[idx] = xr; resOutP[idx] = f2bf(xr);
    } else if (resOutP) {                        // layer2: x3 = h2 + x2 (pack only)
      resOutP[idx] = f2bf(hnew + resIn[idx]);
    }
  }
}

// ---------------- persistent recurrence kernel ----------------
__global__ void __launch_bounds__(64) k_gru_persist(KP P) {
  __shared__ __attribute__((aligned(32))) unsigned sW[L_TOT * 256];  // 140KB weight cache
  __shared__ __attribute__((aligned(16))) float    sB[224];          // 3x64 gate biases + 16 bd

  const int wg = blockIdx.x;
  const int lane = threadIdx.x & 31;
  const int mt = threadIdx.x >> 5;   // mtile 0/1
  const int jt = wg;                 // jtile 0..15

  // ---- one-time fill of the LDS weight cache (per-jtile blocks are contiguous) ----
  {
    auto cp = [&](const unsigned* g, int dstLine, int lines) {
      const uint4* src = (const uint4*)g;
      uint4* dst = (uint4*)(sW + (dstLine << 8));
      int n = lines * 64;                        // uint4 count (line = 256 u32)
      for (int i = threadIdx.x; i < n; i += 64) dst[i] = src[i];
    };
    cp(P.zf0  + (size_t)jt * 12 * 256, L_Z0, 12);
    cp(P.rf0  + (size_t)jt * 12 * 256, L_R0, 12);
    cp(P.xnf0 + (size_t)jt * 4  * 256, L_XN0, 4);
    cp(P.hnf0 + (size_t)jt * 8  * 256, L_HN0, 8);
    cp(P.zf1  + (size_t)jt * 16 * 256, L_Z1, 16);
    cp(P.rf1  + (size_t)jt * 16 * 256, L_R1, 16);
    cp(P.xnf1 + (size_t)jt * 8  * 256, L_XN1, 8);
    cp(P.hnf1 + (size_t)jt * 8  * 256, L_HN1, 8);
    cp(P.zf2  + (size_t)jt * 16 * 256, L_Z2, 16);
    cp(P.rf2  + (size_t)jt * 16 * 256, L_R2, 16);
    cp(P.xnf2 + (size_t)jt * 8  * 256, L_XN2, 8);
    cp(P.hnf2 + (size_t)jt * 8  * 256, L_HN2, 8);
    if (jt < 8) cp(P.dfrag + (size_t)jt * 8 * 256, L_DEN, 8);
    const float* bzrs[3] = { P.bzr0, P.bzr1, P.bzr2 };
    const float* bxns[3] = { P.bxn0, P.bxn1, P.bxn2 };
    const float* bhns[3] = { P.bhn0, P.bhn1, P.bhn2 };
    for (int i = threadIdx.x; i < 224; i += 64) {
      if (i < 192) {
        int l = i >> 6, c = i & 63;
        int gcol = jt * 16 + (c & 15);
        float v;
        switch (c >> 4) {
          case 0:  v = bzrs[l][gcol];        break;   // bz
          case 1:  v = bzrs[l][U_ + gcol];   break;   // br
          case 2:  v = bxns[l][gcol];        break;   // bx
          default: v = bhns[l][gcol];        break;   // bh
        }
        sB[i] = v;
      } else if (i < 208) {
        sB[i] = (jt < 8) ? P.bd[jt * 16 + (i - 192)] : 0.0f;
      } else sB[i] = 0.0f;
    }
    __syncthreads();
  }

  unsigned ep = 0;
  for (int t = 0; t < NSTEP; ++t) {
    const int wp = t & 1, rp = wp ^ 1;
    const int col = jt * 16 + (lane & 15);
    // prefetch next step's warmup inputs (128B per thread covers the 8KB slice)
    if (t + 1 < T_IN_)
      __builtin_prefetch(P.xin + (size_t)(t + 1) * B_ * D_ + threadIdx.x * 64, 0, 1);
    // ---- layer 0: x = inputs+noise (warmup) or previous pred (AR) ----
    const unsigned short* x0 = (t < T_IN_) ? (P.xin + (size_t)t * B_ * D_) : P.predp;
    gru_layer(mt, lane, x0, D_, P.h0p[rp], 4,
              sW, L_Z0, L_R0, L_XN0, L_HN0, sB + 0, col,
              P.h0F, P.h0p[wp], nullptr, nullptr, nullptr);
    gbar(P.bcnt, P.bgen, ep);
    // ---- layer 1: x1 = h0'(t); residual x2 = h1 + x1 ----
    gru_layer(mt, lane, P.h0p[wp], U_, P.h1p[rp], 8,
              sW, L_Z1, L_R1, L_XN1, L_HN1, sB + 64, col,
              P.h1F, P.h1p[wp], P.h0F, P.x2f, P.x2p);
    gbar(P.bcnt, P.bgen, ep);
    // ---- layer 2: x = x2; residual x3 = h2 + x2 ----
    gru_layer(mt, lane, P.x2p, U_, P.h2p[rp], 8,
              sW, L_Z2, L_R2, L_XN2, L_HN2, sB + 128, col,
              P.h2F, P.h2p[wp], P.x2f, nullptr, P.x3p);
    gbar(P.bcnt, P.bgen, ep);
    // ---- dense: pred = x3 @ Wd + bd (only after warmup; 8 jtiles) ----
    if (t >= T_IN_ - 1 && wg < 8) {
      const int hi = lane >> 4, nl = lane & 15, m0 = mt * 16;
      v8f acc = {};
      for (int kc = 0; kc < 8; ++kc)
        acc = wmma_bf16(afrag(P.x3p, U_, m0 + nl, kc, hi),
                        bfrag_s(sW, L_DEN, kc, lane), acc);
      const int dcol = jt * 16 + nl;
      const float bias = sB[192 + nl];
      const int tout = t - (T_IN_ - 1);
#pragma unroll
      for (int r = 0; r < 8; ++r) {
        int m = m0 + r + 8 * hi;                 // m == batch index
        float v = acc[r] + bias;
        P.out[((size_t)m * T_OUT_ + tout) * D_ + dcol] = v;
        P.predp[m * D_ + dcol] = f2bf(v);        // feeds next AR step
      }
    }
    gbar(P.bcnt, P.bgen, ep);
  }
}

// ---------------- host launcher ----------------
extern "C" void kernel_launch(void* const* d_in, const int* in_sizes, int n_in,
                              void* d_out, int out_size, void* d_ws, size_t ws_size,
                              hipStream_t stream) {
  const float* inputs = (const float*)d_in[0];
  const float* noise  = (const float*)d_in[1];
  const float* h_init = (const float*)d_in[2];
  const float* W0 = (const float*)d_in[3];  const float* U0 = (const float*)d_in[4];  const float* b0 = (const float*)d_in[5];
  const float* W1 = (const float*)d_in[6];  const float* U1 = (const float*)d_in[7];  const float* b1 = (const float*)d_in[8];
  const float* W2 = (const float*)d_in[9];  const float* U2 = (const float*)d_in[10]; const float* b2 = (const float*)d_in[11];
  const float* Wd = (const float*)d_in[12]; const float* bd = (const float*)d_in[13];

  char* base = (char*)d_ws; size_t off = 0;
  auto alloc = [&](size_t bytes) -> void* {
    void* r = base + off; off = (off + bytes + 255) & ~(size_t)255; return r;
  };

  KP P;
  P.xin  = (unsigned short*)alloc((size_t)T_IN_ * B_ * D_ * 2);
  P.zf0  = (unsigned*)alloc(16 * 12 * 256 * 4); P.rf0 = (unsigned*)alloc(16 * 12 * 256 * 4);
  P.xnf0 = (unsigned*)alloc(16 * 4  * 256 * 4); P.hnf0 = (unsigned*)alloc(16 * 8 * 256 * 4);
  P.zf1  = (unsigned*)alloc(16 * 16 * 256 * 4); P.rf1 = (unsigned*)alloc(16 * 16 * 256 * 4);
  P.xnf1 = (unsigned*)alloc(16 * 8  * 256 * 4); P.hnf1 = (unsigned*)alloc(16 * 8 * 256 * 4);
  P.zf2  = (unsigned*)alloc(16 * 16 * 256 * 4); P.rf2 = (unsigned*)alloc(16 * 16 * 256 * 4);
  P.xnf2 = (unsigned*)alloc(16 * 8  * 256 * 4); P.hnf2 = (unsigned*)alloc(16 * 8 * 256 * 4);
  P.dfrag = (unsigned*)alloc(8 * 8 * 256 * 4);
  P.bzr0 = (float*)alloc(512 * 4); P.bxn0 = (float*)alloc(256 * 4); P.bhn0 = (float*)alloc(256 * 4);
  P.bzr1 = (float*)alloc(512 * 4); P.bxn1 = (float*)alloc(256 * 4); P.bhn1 = (float*)alloc(256 * 4);
  P.bzr2 = (float*)alloc(512 * 4); P.bxn2 = (float*)alloc(256 * 4); P.bhn2 = (float*)alloc(256 * 4);
  P.bd   = (float*)alloc(D_ * 4);
  P.h0F  = (float*)alloc(B_ * U_ * 4); P.h1F = (float*)alloc(B_ * U_ * 4);
  P.h2F  = (float*)alloc(B_ * U_ * 4); P.x2f = (float*)alloc(B_ * U_ * 4);
  for (int pb = 0; pb < 2; ++pb) {
    P.h0p[pb] = (unsigned short*)alloc(B_ * U_ * 2);
    P.h1p[pb] = (unsigned short*)alloc(B_ * U_ * 2);
    P.h2p[pb] = (unsigned short*)alloc(B_ * U_ * 2);
  }
  P.x2p   = (unsigned short*)alloc(B_ * U_ * 2);
  P.x3p   = (unsigned short*)alloc(B_ * U_ * 2);
  P.predp = (unsigned short*)alloc(B_ * D_ * 2);
  P.bcnt  = (unsigned*)alloc(64);
  P.bgen  = (unsigned*)alloc(64);
  P.out   = (float*)d_out;

  // ---- prep: pack inputs, weights, biases, states (rebuilt every call) ----
  int totX = T_IN_ * B_ * D_;
  k_pack_xin<<<(totX + 255) / 256, 256, 0, stream>>>(inputs, noise, P.xin, totX);

  int tot0 = 2 * 16 * 12 * 256 + 16 * 4 * 256 + 16 * 8 * 256;   // 147456
  k_pack_gru<<<(tot0 + 255) / 256, 256, 0, stream>>>(W0, U0, 128, P.zf0, P.rf0, P.xnf0, P.hnf0, tot0);
  int tot12 = 2 * 16 * 16 * 256 + 16 * 8 * 256 + 16 * 8 * 256;  // 196608
  k_pack_gru<<<(tot12 + 255) / 256, 256, 0, stream>>>(W1, U1, 256, P.zf1, P.rf1, P.xnf1, P.hnf1, tot12);
  k_pack_gru<<<(tot12 + 255) / 256, 256, 0, stream>>>(W2, U2, 256, P.zf2, P.rf2, P.xnf2, P.hnf2, tot12);
  int totD = 8 * 8 * 256;
  k_pack_dense<<<(totD + 255) / 256, 256, 0, stream>>>(Wd, P.dfrag, totD);

  int totI = 3 * B_ * U_ + 3 * 1024 + D_ + 2;
  k_init<<<(totI + 255) / 256, 256, 0, stream>>>(P, h_init, b0, b1, b2, bd);

  // ---- persistent recurrence: whole 511-step GRU stack in one kernel ----
  k_gru_persist<<<NWG, 64, 0, stream>>>(P);
}
